// MultiHeadAttentionRoPE_10359461118256
// MI455X (gfx1250) — matrix operand order repaired and checked by reading, not compiled, not run on hardware
//
#include <hip/hip_runtime.h>
#include <hip/hip_fp16.h>

typedef __attribute__((ext_vector_type(16))) _Float16   v16h;
typedef __attribute__((ext_vector_type(8)))  float      v8f;
typedef __attribute__((ext_vector_type(8)))  unsigned   v8u;

#define DMODEL 1024
#define NHEADS 16
#define DK     64
#define BB     2
#define TT     2048
#define MTOT   (BB * TT)          // 4096 rows for all projections

union FragU { v8u u; v16h h; };

// ---------------------------------------------------------------------------
// Fragment loaders following CDNA5 ISA 7.12.2 wave32 layouts.
// ---------------------------------------------------------------------------

// A-matrix 16x32 f16: lane(0-15)=row M, halves hold K per spec.
__device__ __forceinline__ v16h load_a_frag(const _Float16* X, int ldx,
                                            int row, int k0, int lane) {
    FragU f;
    const unsigned* p = (const unsigned*)(X + (size_t)row * ldx + k0);
    int khalf = (lane >> 4) * 8;
#pragma unroll
    for (int v = 0; v < 8; ++v) {
        int kk = (v < 4 ? 2 * v : 16 + 2 * (v - 4)) + khalf;   // even
        f.u[v] = p[kk >> 1];
    }
    return f.h;
}

// B-matrix 32x16 f16 where B[k][n] = W[n][k] (W row-major [N][K]).
// lanes 0-15 hold K=0..7,16..23 (2 per VGPR), lanes 16-31 hold K=8..15,24..31.
__device__ __forceinline__ v16h load_b_frag_rowk(const _Float16* W, int ldw,
                                                 int ncol, int k0, int lane) {
    FragU f;
    const unsigned* p = (const unsigned*)(W + (size_t)ncol * ldw + k0);
    int kh = (lane >> 4) * 8;
#pragma unroll
    for (int v = 0; v < 8; ++v) {
        int kk = (v < 4 ? 2 * v : 16 + 2 * (v - 4)) + kh;
        f.u[v] = p[kk >> 1];
    }
    return f.h;
}

__device__ __forceinline__ v8f wmma_f16(v16h a, v16h b, v8f c) {
    return __builtin_amdgcn_wmma_f32_16x16x32_f16(false, a, false, b,
                                                  (short)0, c, false, false);
}

// ---------------------------------------------------------------------------
// f32 -> f16 convert
// ---------------------------------------------------------------------------
__global__ void cvt_f32_f16(const float* __restrict__ in,
                            _Float16* __restrict__ out, int n) {
    int i = blockIdx.x * blockDim.x + threadIdx.x;
    if (i < n) out[i] = (_Float16)in[i];
}

// ---------------------------------------------------------------------------
// GEMM  out = X @ W^T.  X: [M,K] f16, W: [N,K] f16.
// Each wave owns a 32x32 output tile: 2 A-frags x 2 B-frags -> 4 WMMAs per
// K-step of 32 (2x the WMMA:load ratio of a 16x16 tile).
// mode 0: f32 store row-major [M,N]          (output projection)
// mode 1: f16 store to [B,H,T,DK] with RoPE  (Q,K projections)
// mode 2: f16 store to [B,H,DK,T] transposed (V projection)
// 8 waves / block, each owning one 32-wide N strip.
// ---------------------------------------------------------------------------
__global__ void __launch_bounds__(256)
gemm_xwT(const _Float16* __restrict__ X,
         const _Float16* __restrict__ W,
         int M, int N, int K,
         float* __restrict__ outF,
         _Float16* __restrict__ outH,
         int mode) {
    int lane  = threadIdx.x & 31;
    int wave  = threadIdx.x >> 5;
    int tileM = blockIdx.x;                   // 32 rows
    int tileN = blockIdx.y * 8 + wave;        // 32 cols

    int row0  = tileM * 32 + (lane & 15);
    int row1  = row0 + 16;
    int ncol0 = tileN * 32 + (lane & 15);
    int ncol1 = ncol0 + 16;

    v8f acc00 = {}, acc01 = {}, acc10 = {}, acc11 = {};
    for (int k0 = 0; k0 < K; k0 += 32) {
        if (k0 + 32 < K) {   // speculative prefetch of next K slice
            __builtin_prefetch(X + (size_t)row0  * K + k0 + 32, 0, 3);
            __builtin_prefetch(X + (size_t)row1  * K + k0 + 32, 0, 3);
            __builtin_prefetch(W + (size_t)ncol0 * K + k0 + 32, 0, 3);
            __builtin_prefetch(W + (size_t)ncol1 * K + k0 + 32, 0, 3);
        }
        v16h a0 = load_a_frag(X, K, row0, k0, lane);
        v16h a1 = load_a_frag(X, K, row1, k0, lane);
        v16h b0 = load_b_frag_rowk(W, K, ncol0, k0, lane);
        v16h b1 = load_b_frag_rowk(W, K, ncol1, k0, lane);
        acc00 = wmma_f16(a0, b0, acc00);
        acc01 = wmma_f16(a0, b1, acc01);
        acc10 = wmma_f16(a1, b0, acc10);
        acc11 = wmma_f16(a1, b1, acc11);
    }

    v8f accs[2][2] = {{acc00, acc01}, {acc10, acc11}};
    int rbase = (lane >> 4) * 8;              // C/D layout: M = r + 8*(lane>=16)

#pragma unroll
    for (int mi = 0; mi < 2; ++mi) {
#pragma unroll
        for (int ni = 0; ni < 2; ++ni) {
            v8f acc  = accs[mi][ni];
            int colN = tileN * 32 + ni * 16 + (lane & 15);
            if (mode == 0) {
#pragma unroll
                for (int r = 0; r < 8; ++r) {
                    int m = tileM * 32 + mi * 16 + rbase + r;
                    outF[(size_t)m * N + colN] = acc[r];
                }
            } else {
                int h = colN >> 6;            // head
                int d = colN & 63;            // dim within head
                bool evn = (d & 1) == 0;
                float inv_freq = powf(10000.0f, -(float)(d & ~1) / 64.0f);
#pragma unroll
                for (int r = 0; r < 8; ++r) {
                    int m = tileM * 32 + mi * 16 + rbase + r;
                    int b = m / TT;
                    int t = m - b * TT;
                    float val = acc[r];
                    if (mode == 1) {
                        float part = __shfl_xor(val, 1);   // pair feature d^1
                        float x1 = evn ? val  : part;
                        float x2 = evn ? part : val;
                        float ang = (float)t * inv_freq;
                        float c = cosf(ang), s = sinf(ang);
                        val = evn ? (x1 * c - x2 * s) : (x1 * s + x2 * c);
                    }
                    size_t off = (mode == 1)
                        ? (((size_t)b * NHEADS + h) * TT + t) * DK + d   // [B,H,T,DK]
                        : (((size_t)b * NHEADS + h) * DK + d) * TT + t;  // [B,H,DK,T]
                    outH[off] = (_Float16)val;
                }
            }
        }
    }
}

// ---------------------------------------------------------------------------
// Flash attention: one wave per (b,h, TWO 16-row q tiles).  K/V fragments
// are loaded once and reused by both q tiles: 16 WMMAs per 8 fragment loads
// per 32-key step.  Online softmax, P relaid C->A layout through LDS.
// Qh/Kh: [B,H,T,DK] f16.  Vt: [B,H,DK,T] f16.  Oh: [B*T, DMODEL] f16.
// ---------------------------------------------------------------------------
__global__ void __launch_bounds__(32)
flash_attn(const _Float16* __restrict__ Qh,
           const _Float16* __restrict__ Kh,
           const _Float16* __restrict__ Vt,
           _Float16* __restrict__ Oh) {
    __shared__ _Float16 Pbuf[2][16][32];

    int lane = threadIdx.x;
    int qp   = blockIdx.x;                     // q tile pair: 0..63
    int bh   = blockIdx.y;                     // b*16+h: 0..31
    int rbase = (lane >> 4) * 8;

    const _Float16* Qp  = Qh + (size_t)bh * TT * DK;
    const _Float16* Kp  = Kh + (size_t)bh * TT * DK;
    const _Float16* Vtp = Vt + (size_t)bh * DK * TT;

    v16h qa[2][2];
#pragma unroll
    for (int ti = 0; ti < 2; ++ti) {
        int qrow = (qp * 2 + ti) * 16 + (lane & 15);
        qa[ti][0] = load_a_frag(Qp, DK, qrow, 0,  lane);   // d = 0..31
        qa[ti][1] = load_a_frag(Qp, DK, qrow, 32, lane);   // d = 32..63
    }

    float m_i[2][8], l_i[2][8];
    v8f accO[2][4];
#pragma unroll
    for (int ti = 0; ti < 2; ++ti) {
#pragma unroll
        for (int r = 0; r < 8; ++r) { m_i[ti][r] = -INFINITY; l_i[ti][r] = 0.0f; }
#pragma unroll
        for (int dt = 0; dt < 4; ++dt) accO[ti][dt] = (v8f){};
    }

    const float scale = 0.125f;                // 1/sqrt(64)

    for (int j = 0; j < TT; j += 32) {
        // ---- shared K fragments for keys j..j+31
        v16h kb00 = load_b_frag_rowk(Kp, DK, j      + (lane & 15), 0,  lane);
        v16h kb01 = load_b_frag_rowk(Kp, DK, j      + (lane & 15), 32, lane);
        v16h kb10 = load_b_frag_rowk(Kp, DK, j + 16 + (lane & 15), 0,  lane);
        v16h kb11 = load_b_frag_rowk(Kp, DK, j + 16 + (lane & 15), 32, lane);

#pragma unroll
        for (int ti = 0; ti < 2; ++ti) {
            v8f s0 = {}, s1 = {};
            s0 = wmma_f16(qa[ti][0], kb00, s0);
            s0 = wmma_f16(qa[ti][1], kb01, s0);
            s1 = wmma_f16(qa[ti][0], kb10, s1);
            s1 = wmma_f16(qa[ti][1], kb11, s1);

            // ---- online softmax update per row
#pragma unroll
            for (int r = 0; r < 8; ++r) {
                float a0 = s0[r] * scale;
                float a1 = s1[r] * scale;
                float mx = fmaxf(a0, a1);
#pragma unroll
                for (int off = 1; off < 16; off <<= 1)
                    mx = fmaxf(mx, __shfl_xor(mx, off));
                float mnew  = fmaxf(m_i[ti][r], mx);
                float alpha = __expf(m_i[ti][r] - mnew);
                m_i[ti][r] = mnew;
                float p0 = __expf(a0 - mnew);
                float p1 = __expf(a1 - mnew);
                float rs = p0 + p1;
#pragma unroll
                for (int off = 1; off < 16; off <<= 1)
                    rs += __shfl_xor(rs, off);
                l_i[ti][r] = l_i[ti][r] * alpha + rs;
#pragma unroll
                for (int dt = 0; dt < 4; ++dt) accO[ti][dt][r] *= alpha;
                Pbuf[ti][rbase + r][lane & 15]        = (_Float16)p0;
                Pbuf[ti][rbase + r][16 + (lane & 15)] = (_Float16)p1;
            }
        }
        __syncthreads();   // single-wave WG: compiler fence, S_NOP on HW

        // ---- P (16x32 f16 A frags from LDS) @ V (32x64), V frag shared
        v16h pa0 = load_a_frag(&Pbuf[0][0][0], 32, lane & 15, 0, lane);
        v16h pa1 = load_a_frag(&Pbuf[1][0][0], 32, lane & 15, 0, lane);
#pragma unroll
        for (int dt = 0; dt < 4; ++dt) {
            v16h vb = load_b_frag_rowk(Vtp, TT, dt * 16 + (lane & 15), j, lane);
            accO[0][dt] = wmma_f16(pa0, vb, accO[0][dt]);
            accO[1][dt] = wmma_f16(pa1, vb, accO[1][dt]);
        }
        __syncthreads();
    }

    // ---- normalize and scatter to [B*T, DMODEL] f16
    int b = bh >> 4, h = bh & 15;
#pragma unroll
    for (int ti = 0; ti < 2; ++ti) {
#pragma unroll
        for (int dt = 0; dt < 4; ++dt) {
#pragma unroll
            for (int r = 0; r < 8; ++r) {
                int q = (qp * 2 + ti) * 16 + rbase + r;
                int d = dt * 16 + (lane & 15);
                float o = accO[ti][dt][r] / l_i[ti][r];
                Oh[((size_t)b * TT + q) * DMODEL + h * DK + d] = (_Float16)o;
            }
        }
    }
}

// ---------------------------------------------------------------------------
// Launch
// ---------------------------------------------------------------------------
extern "C" void kernel_launch(void* const* d_in, const int* in_sizes, int n_in,
                              void* d_out, int out_size, void* d_ws, size_t ws_size,
                              hipStream_t stream) {
    (void)in_sizes; (void)n_in; (void)out_size; (void)ws_size;

    const float* q  = (const float*)d_in[0];
    const float* k  = (const float*)d_in[1];
    const float* v  = (const float*)d_in[2];
    const float* Wq = (const float*)d_in[3];
    const float* Wk = (const float*)d_in[4];
    const float* Wv = (const float*)d_in[5];
    const float* Wo = (const float*)d_in[6];
    float* out = (float*)d_out;

    const size_t NX = (size_t)MTOT * DMODEL;   // 4Mi elements
    const size_t NW = (size_t)DMODEL * DMODEL; // 1Mi elements

    _Float16* ws  = (_Float16*)d_ws;
    _Float16* qh  = ws;                 // 4Mi
    _Float16* kh  = qh  + NX;           // 4Mi
    _Float16* vh  = kh  + NX;           // 4Mi
    _Float16* wqh = vh  + NX;           // 1Mi
    _Float16* wkh = wqh + NW;
    _Float16* wvh = wkh + NW;
    _Float16* woh = wvh + NW;
    _Float16* Qr  = woh + NW;           // [B,H,T,DK]  4Mi
    _Float16* Kr  = Qr  + NX;           // [B,H,T,DK]  4Mi
    _Float16* Vtr = Kr  + NX;           // [B,H,DK,T]  4Mi
    _Float16* Ohb = Vtr + NX;           // [B*T, DMODEL] 4Mi

    // converts
    {
        int blk = 256;
        cvt_f32_f16<<<(int)((NX + blk - 1) / blk), blk, 0, stream>>>(q,  qh, (int)NX);
        cvt_f32_f16<<<(int)((NX + blk - 1) / blk), blk, 0, stream>>>(k,  kh, (int)NX);
        cvt_f32_f16<<<(int)((NX + blk - 1) / blk), blk, 0, stream>>>(v,  vh, (int)NX);
        cvt_f32_f16<<<(int)((NW + blk - 1) / blk), blk, 0, stream>>>(Wq, wqh, (int)NW);
        cvt_f32_f16<<<(int)((NW + blk - 1) / blk), blk, 0, stream>>>(Wk, wkh, (int)NW);
        cvt_f32_f16<<<(int)((NW + blk - 1) / blk), blk, 0, stream>>>(Wv, wvh, (int)NW);
        cvt_f32_f16<<<(int)((NW + blk - 1) / blk), blk, 0, stream>>>(Wo, woh, (int)NW);
    }

    // projections: 32x32 tile per wave, 8 waves/block
    dim3 gblk(256);
    dim3 ggrd(MTOT / 32, DMODEL / 32 / 8);
    gemm_xwT<<<ggrd, gblk, 0, stream>>>(qh, wqh, MTOT, DMODEL, DMODEL,
                                        nullptr, Qr,  /*mode=*/1);
    gemm_xwT<<<ggrd, gblk, 0, stream>>>(kh, wkh, MTOT, DMODEL, DMODEL,
                                        nullptr, Kr,  /*mode=*/1);
    gemm_xwT<<<ggrd, gblk, 0, stream>>>(vh, wvh, MTOT, DMODEL, DMODEL,
                                        nullptr, Vtr, /*mode=*/2);

    // attention: one wave per (2 q-tiles, b*h)
    flash_attn<<<dim3(TT / 32, BB * NHEADS), dim3(32), 0, stream>>>(Qr, Kr, Vtr, Ohb);

    // output projection -> f32
    gemm_xwT<<<ggrd, gblk, 0, stream>>>(Ohb, woh, MTOT, DMODEL, DMODEL,
                                        out, nullptr, /*mode=*/0);
}
